// WavLMSelfAttention_2731599200281
// MI455X (gfx1250) — matrix operand order repaired and checked by reading, not compiled, not run on hardware
//
#include <hip/hip_runtime.h>
#include <hip/hip_bf16.h>

#define EMBED 768
#define NH    12
#define HD    64
#define BB    8
#define SS    1024

typedef __attribute__((ext_vector_type(8)))  float  v8f;
typedef __attribute__((ext_vector_type(8)))  __bf16 v8bf;
typedef __attribute__((ext_vector_type(16))) __bf16 v16bf;

#if defined(__gfx1250__) && __has_builtin(__builtin_amdgcn_tensor_load_to_lds) && __has_builtin(__builtin_amdgcn_s_wait_tensorcnt)
#define USE_TDM 1
#else
#define USE_TDM 0
#endif

union BFrag { v16bf v; v8bf h[2]; };

// A-matrix 16x32 bf16 fragment: lanes 0-15 row M=lane (K base 0), lanes 16-31 row M=lane-16 (K base 8);
// per lane: K = base+0..7 and base+16..23  -> two contiguous 16B chunks.
__device__ __forceinline__ v16bf load_frag_a(const __bf16* base, int ld, int lane) {
    const __bf16* p = base + (size_t)(lane & 15) * ld + ((lane >> 4) << 3);
    BFrag u;
    u.h[0] = *(const v8bf*)(p);
    u.h[1] = *(const v8bf*)(p + 16);
    return u.v;
}
// B-matrix 32x16 bf16 fragment: lanes 0-15 column N=lane (K base 0), lanes 16-31 column N=lane-16 (K base 8);
// per lane: K = base+0..7 and base+16..23 (strided by ld).
__device__ __forceinline__ v16bf load_frag_b(const __bf16* base, int ld, int lane) {
    const __bf16* p = base + (size_t)((lane >> 4) << 3) * ld + (lane & 15);
    v16bf v;
    #pragma unroll
    for (int i = 0; i < 8; ++i) {
        v[i] = p[(size_t)i * ld];
        v[8 + i] = p[(size_t)(16 + i) * ld];
    }
    return v;
}
__device__ __forceinline__ v8f wmma_bf16(v16bf a, v16bf b, v8f c) {
    return __builtin_amdgcn_wmma_f32_16x16x32_bf16(false, a, false, b, (short)0, c, false, false);
}

#if USE_TDM
typedef unsigned int u32x4 __attribute__((ext_vector_type(4)));
typedef int          i32x4 __attribute__((ext_vector_type(4)));
typedef int          i32x8 __attribute__((ext_vector_type(8)));

// TDM: DMA a 32-row x 64-col bf16 tile of a (768,768) row-major tensor into LDS.
// D# group0: [1:0]=count=1 | [63:32]=lds_addr | [120:64]=global_addr | [127:126]=type=2
// D# group1: data_size=1(2B), tensor_dim0=768 @ bits 79:48, tensor_dim1=768 @ 111:80,
//            tile_dim0=64 @ 127:112, tile_dim1=32 @ 143:128, dim0_stride=768 @ 207:160
__device__ __forceinline__ void tdm_load_b_tile(const __bf16* gptr, __bf16* lptr) {
    unsigned long long ga = (unsigned long long)(const void*)gptr;
    unsigned int la = (unsigned int)(unsigned long long)(void*)lptr; // flat LDS addr low 32 = LDS offset
    u32x4 g0;
    g0[0] = 1u;
    g0[1] = la;
    g0[2] = (unsigned int)ga;
    g0[3] = (unsigned int)((ga >> 32) & 0x1FFFFFFull) | (2u << 30);
    i32x8 g1;
    g1[0] = 0x00010000;            // workgroup_mask=0, data_size=1 (2 bytes)
    g1[1] = (int)(768u << 16);     // tensor_dim0[15:0] in [63:48]
    g1[2] = (int)(768u << 16);     // tensor_dim0 hi=0, tensor_dim1[15:0] in [95:80]
    g1[3] = (int)(64u << 16);      // tensor_dim1 hi=0, tile_dim0=64
    g1[4] = 32;                    // tile_dim1=32, tile_dim2=0
    g1[5] = 768;                   // tensor_dim0_stride lo
    g1[6] = 0;                     // stride hi, dim1_stride lo16
    g1[7] = 0;
    i32x4 gz = {0, 0, 0, 0};
#if __clang_major__ >= 23
    i32x8 gz8 = {0, 0, 0, 0, 0, 0, 0, 0};
    __builtin_amdgcn_tensor_load_to_lds(g0, g1, gz, gz, gz8, 0);
#else
    __builtin_amdgcn_tensor_load_to_lds(g0, g1, gz, gz, 0);
#endif
}
#endif

// ---------------- prep kernels ----------------

__global__ void cvt_f32_bf16(const float* __restrict__ in, __bf16* __restrict__ out, int n) {
    int i = blockIdx.x * blockDim.x + threadIdx.x;
    if (i < n) out[i] = (__bf16)in[i];
}

// biasTab[h][delta + S-1] = rel_embed[bucket(delta)][h]
__global__ void bias_table_kernel(const float* __restrict__ rel, float* __restrict__ btab) {
    int idx = blockIdx.x * blockDim.x + threadIdx.x;
    const int W = 2 * SS - 1;
    if (idx >= NH * W) return;
    int h = idx / W;
    int rp = (idx % W) - (SS - 1);
    int bucket = (rp > 0) ? 16 : 0;
    int arp = rp < 0 ? -rp : rp;
    if (arp < 8) {
        bucket += arp;
    } else {
        int large = 8 + (int)(__logf((float)arp / 8.0f) / __logf(128.0f / 8.0f) * 8.0f);
        bucket += (large < 15) ? large : 15;
    }
    btab[idx] = rel[bucket * NH + h];
}

// gate[(b*NH+h)*S + s] = g0*(g1*gc[h]-1)+2  with g = sigmoid of 4-group sums of ql@gru_W+gru_b
__global__ void gate_kernel(const float* __restrict__ q, const float* __restrict__ gruW,
                            const float* __restrict__ gruB, const float* __restrict__ gruC,
                            float* __restrict__ gate) {
    int idx = blockIdx.x * blockDim.x + threadIdx.x;
    if (idx >= BB * SS * NH) return;
    int h = idx % NH;
    int s = (idx / NH) % SS;
    int b = idx / (NH * SS);
    const float* x = q + ((size_t)(b * SS + s)) * EMBED + h * HD;
    float acc[8] = {0, 0, 0, 0, 0, 0, 0, 0};
    for (int i = 0; i < HD; ++i) {
        float xv = x[i];
        #pragma unroll
        for (int j = 0; j < 8; ++j) acc[j] += xv * gruW[i * 8 + j];
    }
    float s0 = acc[0] + acc[1] + acc[2] + acc[3] + gruB[0] + gruB[1] + gruB[2] + gruB[3];
    float s1 = acc[4] + acc[5] + acc[6] + acc[7] + gruB[4] + gruB[5] + gruB[6] + gruB[7];
    float g0 = 1.0f / (1.0f + __expf(-s0));
    float g1 = 1.0f / (1.0f + __expf(-s1));
    gate[((size_t)(b * NH + h)) * SS + s] = g0 * (g1 * gruC[h] - 1.0f) + 2.0f;
}

// ---------------- WMMA GEMM: C(M,768) = A(M,768)*Bw(768,768) + bias ----------------
// Block: 8 waves x (32M x 64N) = 256M x 64N. B tile (32K x 64N) staged in LDS via TDM
// (double-buffered), shared by all 8 waves. A fragments stream from global.
// mode 0: out bf16 (B,H,S,D) scaled   (Q)
// mode 1: out bf16 (B,H,D,S)          (K transposed)
// mode 2: out bf16 (B,H,S,D)          (V)
// mode 3: out fp32 row-major (M,768)  (final projection)
__global__ __launch_bounds__(256) void gemm_wmma(
    const __bf16* __restrict__ A, const __bf16* __restrict__ Bw,
    const float* __restrict__ bias, __bf16* __restrict__ obf,
    float* __restrict__ ofp, int mode, float scale) {
    const int tid = threadIdx.x, wave = tid >> 5, lane = tid & 31;
    const int m0 = (blockIdx.y * 8 + wave) * 32;
    const int n0 = blockIdx.x * 64;
    const int NK = EMBED / 32; // 24 K-steps

    v8f acc[2][4];
    #pragma unroll
    for (int mi = 0; mi < 2; ++mi)
        #pragma unroll
        for (int n = 0; n < 4; ++n) acc[mi][n] = (v8f){};

#if USE_TDM
    __shared__ __align__(16) __bf16 bstage[2][32 * 64];
    if (wave == 0) tdm_load_b_tile(Bw + n0, &bstage[0][0]);
    for (int kt = 0; kt < NK; ++kt) {
        const int k = kt * 32;
        if (wave == 0) {
            if (kt + 1 < NK) {
                tdm_load_b_tile(Bw + (size_t)(k + 32) * EMBED + n0, &bstage[(kt + 1) & 1][0]);
                __builtin_amdgcn_s_wait_tensorcnt(1);
            } else {
                __builtin_amdgcn_s_wait_tensorcnt(0);
            }
        }
        __syncthreads();
        const __bf16* bp = &bstage[kt & 1][0]; // 32x64 row-major, ld=64
        #pragma unroll
        for (int mi = 0; mi < 2; ++mi) {
            const __bf16* ap = A + (size_t)(m0 + mi * 16) * EMBED + k;
            __builtin_prefetch(ap + 32, 0, 1); // hint next A chunk
            v16bf a = load_frag_a(ap, EMBED, lane);
            acc[mi][0] = wmma_bf16(a, load_frag_b(bp + 0,  64, lane), acc[mi][0]);
            acc[mi][1] = wmma_bf16(a, load_frag_b(bp + 16, 64, lane), acc[mi][1]);
            acc[mi][2] = wmma_bf16(a, load_frag_b(bp + 32, 64, lane), acc[mi][2]);
            acc[mi][3] = wmma_bf16(a, load_frag_b(bp + 48, 64, lane), acc[mi][3]);
        }
        __syncthreads();
    }
#else
    for (int k = 0; k < EMBED; k += 32) {
        const __bf16* bp = Bw + (size_t)k * EMBED + n0;
        v16bf b0 = load_frag_b(bp + 0,  EMBED, lane);
        v16bf b1 = load_frag_b(bp + 16, EMBED, lane);
        v16bf b2 = load_frag_b(bp + 32, EMBED, lane);
        v16bf b3 = load_frag_b(bp + 48, EMBED, lane);
        #pragma unroll
        for (int mi = 0; mi < 2; ++mi) {
            v16bf a = load_frag_a(A + (size_t)(m0 + mi * 16) * EMBED + k, EMBED, lane);
            acc[mi][0] = wmma_bf16(a, b0, acc[mi][0]);
            acc[mi][1] = wmma_bf16(a, b1, acc[mi][1]);
            acc[mi][2] = wmma_bf16(a, b2, acc[mi][2]);
            acc[mi][3] = wmma_bf16(a, b3, acc[mi][3]);
        }
    }
#endif

    const int hlf = lane >> 4, ln = lane & 15;
    #pragma unroll
    for (int mi = 0; mi < 2; ++mi) {
        #pragma unroll
        for (int n = 0; n < 4; ++n) {
            #pragma unroll
            for (int j = 0; j < 8; ++j) {
                int m = m0 + mi * 16 + j + 8 * hlf;
                int col = n0 + n * 16 + ln;
                float val = (acc[mi][n][j] + bias[col]) * scale;
                if (mode == 3) {
                    ofp[(size_t)m * EMBED + col] = val;
                } else {
                    int bb = m >> 10, s = m & (SS - 1);
                    int hh = col >> 6, d = col & 63;
                    if (mode == 1)
                        obf[(((size_t)(bb * NH + hh)) * HD + d) * SS + s] = (__bf16)val;
                    else
                        obf[(((size_t)(bb * NH + hh)) * SS + s) * HD + d] = (__bf16)val;
                }
            }
        }
    }
}

// ---------------- flash attention: block = 16 queries of one (b,h); 8 waves x 128 keys ----------------
__global__ __launch_bounds__(256) void attn_kernel(
    const __bf16* __restrict__ qb, const __bf16* __restrict__ kt,
    const __bf16* __restrict__ vb, const float* __restrict__ gate,
    const float* __restrict__ btab, __bf16* __restrict__ aout) {
    __shared__ __align__(16) __bf16 pstage[8][16 * 32];   // P tiles, per-wave region
    __shared__ float lm[8][16];
    __shared__ float ll[8][16];
    __shared__ float lO[8][16][64];
    __shared__ float bwin[1056];

    const int tid = threadIdx.x;
    const int wave = tid >> 5, lane = tid & 31;
    const int hlf = lane >> 4, ln = lane & 15;
    const int q0 = blockIdx.x * 16;
    const int h = blockIdx.y, b = blockIdx.z;

    // stage relative-position bias window: element (r,key) reads bwin[key - r + 15]
    const int woff = SS - 16 - q0;
    for (int i = tid; i < 1039; i += 256)
        bwin[i] = btab[h * (2 * SS - 1) + i + woff];
    __syncthreads();

    const __bf16* qhead = qb + ((size_t)(b * NH + h) * SS) * HD;
    const __bf16* khead = kt + ((size_t)(b * NH + h) * HD) * SS;
    const __bf16* vhead = vb + ((size_t)(b * NH + h) * SS) * HD;

    v16bf aq0 = load_frag_a(qhead + (size_t)q0 * HD, HD, lane);      // d 0..31
    v16bf aq1 = load_frag_a(qhead + (size_t)q0 * HD + 32, HD, lane); // d 32..63

    float grow[8], m_r[8], l_r[8];
    #pragma unroll
    for (int j = 0; j < 8; ++j) {
        grow[j] = gate[((size_t)(b * NH + h)) * SS + q0 + j + 8 * hlf];
        m_r[j] = -1e30f;
        l_r[j] = 0.0f;
    }
    v8f o0 = {}, o1 = {}, o2 = {}, o3 = {};

    for (int kt32 = 0; kt32 < 4; ++kt32) {
        const int kb32 = wave * 128 + kt32 * 32;
        // scores for two 16-key subtiles (joint max -> K=32 of P per PV step)
        v8f ca = {}, cb = {};
        ca = wmma_bf16(aq0, load_frag_b(khead + kb32,           SS, lane), ca);
        ca = wmma_bf16(aq1, load_frag_b(khead + 32 * SS + kb32, SS, lane), ca);
        cb = wmma_bf16(aq0, load_frag_b(khead + kb32 + 16,           SS, lane), cb);
        cb = wmma_bf16(aq1, load_frag_b(khead + 32 * SS + kb32 + 16, SS, lane), cb);

        #pragma unroll
        for (int j = 0; j < 8; ++j) {
            int r = j + 8 * hlf;                 // row within 16-query tile
            int ka = kb32 + ln, kb = kb32 + 16 + ln;
            float sa = ca[j] + grow[j] * bwin[ka - r + 15];
            float sb = cb[j] + grow[j] * bwin[kb - r + 15];
            // row max across the 16-lane half (xor <= 8 stays within halves)
            float tm = fmaxf(sa, sb);
            tm = fmaxf(tm, __shfl_xor(tm, 1, 32));
            tm = fmaxf(tm, __shfl_xor(tm, 2, 32));
            tm = fmaxf(tm, __shfl_xor(tm, 4, 32));
            tm = fmaxf(tm, __shfl_xor(tm, 8, 32));
            float nm = fmaxf(m_r[j], tm);
            float corr = __expf(m_r[j] - nm);
            float pa = __expf(sa - nm);
            float pb = __expf(sb - nm);
            float ps = pa + pb;
            ps += __shfl_xor(ps, 1, 32);
            ps += __shfl_xor(ps, 2, 32);
            ps += __shfl_xor(ps, 4, 32);
            ps += __shfl_xor(ps, 8, 32);
            l_r[j] = l_r[j] * corr + ps;
            m_r[j] = nm;
            o0[j] *= corr; o1[j] *= corr; o2[j] *= corr; o3[j] *= corr;
            pstage[wave][r * 32 + ln]      = (__bf16)pa;
            pstage[wave][r * 32 + 16 + ln] = (__bf16)pb;
        }
        asm volatile("" ::: "memory"); // keep C-layout stores before A-layout loads (DS is in-order per wave)
        v16bf ap = load_frag_a(&pstage[wave][0], 32, lane);
        const __bf16* vbase = vhead + (size_t)kb32 * HD;
        o0 = wmma_bf16(ap, load_frag_b(vbase + 0,  HD, lane), o0);
        o1 = wmma_bf16(ap, load_frag_b(vbase + 16, HD, lane), o1);
        o2 = wmma_bf16(ap, load_frag_b(vbase + 32, HD, lane), o2);
        o3 = wmma_bf16(ap, load_frag_b(vbase + 48, HD, lane), o3);
        asm volatile("" ::: "memory");
    }

    // publish per-wave partials
    #pragma unroll
    for (int j = 0; j < 8; ++j) {
        int r = j + 8 * hlf;
        if (ln == 0) { lm[wave][r] = m_r[j]; ll[wave][r] = l_r[j]; }
        lO[wave][r][0  + ln] = o0[j];
        lO[wave][r][16 + ln] = o1[j];
        lO[wave][r][32 + ln] = o2[j];
        lO[wave][r][48 + ln] = o3[j];
    }
    __syncthreads();

    // flash merge of 8 key-range partials: thread -> (row, 4 d-elements)
    const int row = tid >> 4, dg = tid & 15;
    float gm = -1e30f;
    #pragma unroll
    for (int wv = 0; wv < 8; ++wv) gm = fmaxf(gm, lm[wv][row]);
    float L = 0.0f, a0 = 0, a1 = 0, a2 = 0, a3 = 0;
    #pragma unroll
    for (int wv = 0; wv < 8; ++wv) {
        float e = __expf(lm[wv][row] - gm);
        L  += e * ll[wv][row];
        a0 += e * lO[wv][row][dg * 4 + 0];
        a1 += e * lO[wv][row][dg * 4 + 1];
        a2 += e * lO[wv][row][dg * 4 + 2];
        a3 += e * lO[wv][row][dg * 4 + 3];
    }
    float inv = 1.0f / L;
    __bf16* op = aout + ((size_t)(b * SS + q0 + row)) * EMBED + h * HD + dg * 4;
    op[0] = (__bf16)(a0 * inv);
    op[1] = (__bf16)(a1 * inv);
    op[2] = (__bf16)(a2 * inv);
    op[3] = (__bf16)(a3 * inv);
}

// ---------------- launch ----------------

extern "C" void kernel_launch(void* const* d_in, const int* in_sizes, int n_in,
                              void* d_out, int out_size, void* d_ws, size_t ws_size,
                              hipStream_t stream) {
    const float* query = (const float*)d_in[0];
    const float* Wq    = (const float*)d_in[1];
    const float* bq    = (const float*)d_in[2];
    const float* Wk    = (const float*)d_in[3];
    const float* bk    = (const float*)d_in[4];
    const float* Wv    = (const float*)d_in[5];
    const float* bv    = (const float*)d_in[6];
    const float* Wo    = (const float*)d_in[7];
    const float* bo    = (const float*)d_in[8];
    const float* rel   = (const float*)d_in[9];
    const float* gruW  = (const float*)d_in[10];
    const float* gruB  = (const float*)d_in[11];
    const float* gruC  = (const float*)d_in[12];
    float* out = (float*)d_out;

    char* w = (char*)d_ws;
    auto alloc = [&](size_t bytes) -> char* {
        char* p = w;
        w += (bytes + 255) & ~(size_t)255;
        return p;
    };
    const size_t TOK = (size_t)BB * SS * EMBED;   // 6.29M elements
    const size_t WEL = (size_t)EMBED * EMBED;
    __bf16* Qtok = (__bf16*)alloc(TOK * 2);
    __bf16* Wqb  = (__bf16*)alloc(WEL * 2);
    __bf16* Wkb  = (__bf16*)alloc(WEL * 2);
    __bf16* Wvb  = (__bf16*)alloc(WEL * 2);
    __bf16* Wob  = (__bf16*)alloc(WEL * 2);
    __bf16* qbuf = (__bf16*)alloc(TOK * 2);       // (B,H,S,D)
    __bf16* ktb  = (__bf16*)alloc(TOK * 2);       // (B,H,D,S)
    __bf16* vbuf = (__bf16*)alloc(TOK * 2);       // (B,H,S,D)
    __bf16* aout = (__bf16*)alloc(TOK * 2);       // (B,S,E)
    float*  gateb = (float*)alloc((size_t)BB * NH * SS * 4);
    float*  btab  = (float*)alloc((size_t)NH * (2 * SS - 1) * 4);

    // 1) converts + tables
    cvt_f32_bf16<<<(int)((TOK + 255) / 256), 256, 0, stream>>>(query, Qtok, (int)TOK);
    cvt_f32_bf16<<<(int)((WEL + 255) / 256), 256, 0, stream>>>(Wq, Wqb, (int)WEL);
    cvt_f32_bf16<<<(int)((WEL + 255) / 256), 256, 0, stream>>>(Wk, Wkb, (int)WEL);
    cvt_f32_bf16<<<(int)((WEL + 255) / 256), 256, 0, stream>>>(Wv, Wvb, (int)WEL);
    cvt_f32_bf16<<<(int)((WEL + 255) / 256), 256, 0, stream>>>(Wo, Wob, (int)WEL);
    bias_table_kernel<<<(NH * (2 * SS - 1) + 255) / 256, 256, 0, stream>>>(rel, btab);
    gate_kernel<<<(BB * SS * NH + 255) / 256, 256, 0, stream>>>(query, gruW, gruB, gruC, gateb);

    // 2) QKV projections (M=8192, N=768, K=768); block = 256M x 64N
    dim3 ggrid(EMBED / 64, (BB * SS) / 256);
    const float scale = 0.125f; // 1/sqrt(64), applied after bias add as in reference
    gemm_wmma<<<ggrid, 256, 0, stream>>>(Qtok, Wqb, bq, qbuf, nullptr, 0, scale);
    gemm_wmma<<<ggrid, 256, 0, stream>>>(Qtok, Wkb, bk, ktb,  nullptr, 1, 1.0f);
    gemm_wmma<<<ggrid, 256, 0, stream>>>(Qtok, Wvb, bv, vbuf, nullptr, 2, 1.0f);

    // 3) attention
    dim3 agrid(SS / 16, NH, BB);
    attn_kernel<<<agrid, 256, 0, stream>>>(qbuf, ktb, vbuf, gateb, btab, aout);

    // 4) output projection -> fp32 d_out
    gemm_wmma<<<ggrid, 256, 0, stream>>>(aout, Wob, bo, nullptr, out, 3, 1.0f);
}
